// SI_GNN_41111426957465
// MI455X (gfx1250) — matrix operand order repaired and checked by reading, not compiled, not run on hardware
//
#include <hip/hip_runtime.h>
#include <math.h>
#include <stdint.h>

// ---------------------------------------------------------------------------
// SI-GNN forward for MI455X (gfx1250), wave32 + WMMA bf16.
// Matmuls: v_wmma_f32_16x16x32_bf16, f32 accumulate, double-buffered LDS,
// async global->LDS (ASYNCcnt) for pre-converted bf16 weight tiles,
// global_prefetch for streamed activations. GEMM is templated on
// (B layout, B dtype, epilogue, tile shape) so each hot loop is branch-free.
// ---------------------------------------------------------------------------

#define B_     256
#define LOC_   100
#define NS_    50
#define NTOT_  150      // LOC + NS
#define BBD_   64
#define FD_    512
#define FDBB_  576      // FD + BB
#define DIN_   2048
#define SIN_   300

typedef __attribute__((ext_vector_type(16))) __bf16 v16bf;
typedef __attribute__((ext_vector_type(8)))  float  v8f;

union FragBF { v16bf v; uint4 q[2]; };
union Pack8  { uint4 q; unsigned short h[8]; };

__device__ __forceinline__ unsigned short f2b(float f) {
  union { float f; unsigned u; } x; x.f = f;
  unsigned u = x.u;
  unsigned r = u + 0x7FFFu + ((u >> 16) & 1u);      // round-to-nearest-even
  if ((u & 0x7F800000u) == 0x7F800000u) r = u;      // inf/nan: truncate
  return (unsigned short)(r >> 16);
}

// ---------------------------------------------------------------------------
// Batched WMMA GEMM:  C[m,n] = epi( sum_k A[m,k] * B(n,k) + bias[n] )
//   A: f32, row-remappable: arow = a_base + (m/a_rpb)*a_bsr + m%a_rpb
//   B: BNN==0 -> B[N][K] ("x @ W^T"), BNN==1 -> B[K][N]
//      BF32 selects f32 (converted on stage) vs pre-converted bf16 source;
//      bf16 NT full tiles go HBM->LDS via global_load_async_to_lds_b128.
//   C: f32, row-remappable like A, ldc == N.
//   EPI: 0 none, 1 relu(acc+bias), 2 Rsd + (it>0 ? tanh(acc) : 0)
// Block: 256 thr (8 waves, 4(M) x 2(N)); wave tile (BMt/4) x (BNt/2);
// per wave per K-step: WR=BMt/64 A-frags, WC=BNt/32 B-frags, WR*WC WMMAs.
// ---------------------------------------------------------------------------
#define BK 32
#define LDT 40   // LDS row stride in halves (16B aligned, conflict-padded)

template<bool BNN, bool BF32, int EPI, int BMt, int BNt>
__global__ __launch_bounds__(256) void gemm_wmma_bf16(
    const float* __restrict__ A, const void* __restrict__ Bp,
    const float* __restrict__ bias, const float* __restrict__ Rsd,
    const int* __restrict__ itp, float* __restrict__ C,
    int M, int N, int K,
    long long aBS, long long bBS, long long cBS,
    int a_rpb, int a_bsr, int a_base,
    int c_rpb, int c_bsr, int c_base)
{
  constexpr int WR = BMt / 64;          // A fragments per wave
  constexpr int WC = BNt / 32;          // B fragments per wave
  constexpr bool ASYNC_B = (!BNN) && (!BF32);

  __shared__ __align__(16) unsigned short sA[2][BMt * LDT];
  __shared__ __align__(16) unsigned short sB[2][BNt * LDT];

  const int z = blockIdx.z;
  const float* Ab = A + (size_t)z * aBS;
  const float* Bf = (const float*)Bp + (size_t)z * bBS;          // f32 B path
  const unsigned short* Bh = (const unsigned short*)Bp;          // bf16 weights
  float* Cb = C + (size_t)z * cBS;
  const float* Rb = Rsd ? (Rsd + (size_t)z * cBS) : nullptr;

  const int bm = blockIdx.y * BMt, bn = blockIdx.x * BNt;
  const int t = threadIdx.x, w = t >> 5, lane = t & 31;
  const int wm = (w & 3) * (BMt / 4);   // wave row offset within tile
  const int wn = (w >> 2) * (BNt / 2);  // wave col offset within tile
  const int lh = lane >> 4, ln = lane & 15;

  v8f acc[WR][WC];
#pragma unroll
  for (int r = 0; r < WR; ++r)
#pragma unroll
    for (int c = 0; c < WC; ++c)
      acc[r][c] = (v8f){0.f,0.f,0.f,0.f,0.f,0.f,0.f,0.f};

  // ---- tile staging: one (row, 8-col chunk) per thread slot ----
  auto stage = [&](int kt, int buf) {
    const int k0 = kt * BK;
    // A tile: BMt x 32 f32 -> bf16
#pragma unroll
    for (int cc = 0; cc < BMt / 64; ++cc) {
      const int slot = t + cc * 256;
      const int r = slot >> 2, cq = (slot & 3) * 8;
      const int m = bm + r;
      float vals[8];
#pragma unroll
      for (int j = 0; j < 8; ++j) vals[j] = 0.f;
      if (m < M) {
        int ar = m;
        if (a_bsr) ar = a_base + (m / a_rpb) * a_bsr + (m % a_rpb);
        const float* src = Ab + (size_t)ar * K + k0 + cq;
        if (k0 + cq + 8 <= K) {
#pragma unroll
          for (int j = 0; j < 8; ++j) vals[j] = src[j];
          if (k0 + cq + 8 + BK <= K) __builtin_prefetch(src + BK, 0, 0);
        } else {
#pragma unroll
          for (int j = 0; j < 8; ++j) if (k0 + cq + j < K) vals[j] = src[j];
        }
      }
      Pack8 pk;
#pragma unroll
      for (int j = 0; j < 8; ++j) pk.h[j] = f2b(vals[j]);
      *reinterpret_cast<uint4*>(&sA[buf][r * LDT + cq]) = pk.q;
    }
    // B tile into sB[n][k]
    if (!BNN) {
#pragma unroll
      for (int cc = 0; cc < BNt / 64; ++cc) {
        const int slot = t + cc * 256;
        const int n = slot >> 2, cq = (slot & 3) * 8;
        const int gn = bn + n;
        if (!BF32) {
          // bf16 weights: full 16B-aligned chunks go HBM -> LDS directly.
          if (gn < N && (k0 + cq + 8 <= K) && ((K & 7) == 0)) {
            unsigned lds = (unsigned)(uintptr_t)&sB[buf][n * LDT + cq];
            unsigned long long ga =
                (unsigned long long)(uintptr_t)(Bh + (size_t)gn * K + k0 + cq);
            asm volatile("global_load_async_to_lds_b128 %0, %1, off"
                         :: "v"(lds), "v"(ga) : "memory");
          } else {
            Pack8 pk;
#pragma unroll
            for (int j = 0; j < 8; ++j) {
              unsigned short hv = 0;
              if (gn < N && (k0 + cq + j) < K)
                hv = Bh[(size_t)gn * K + k0 + cq + j];
              pk.h[j] = hv;
            }
            *reinterpret_cast<uint4*>(&sB[buf][n * LDT + cq]) = pk.q;
          }
        } else {
          float vals[8];
#pragma unroll
          for (int j = 0; j < 8; ++j) vals[j] = 0.f;
          if (gn < N) {
            const float* src = Bf + (size_t)gn * K + k0 + cq;
            if (k0 + cq + 8 <= K) {
#pragma unroll
              for (int j = 0; j < 8; ++j) vals[j] = src[j];
            } else {
#pragma unroll
              for (int j = 0; j < 8; ++j) if (k0 + cq + j < K) vals[j] = src[j];
            }
          }
          Pack8 pk;
#pragma unroll
          for (int j = 0; j < 8; ++j) pk.h[j] = f2b(vals[j]);
          *reinterpret_cast<uint4*>(&sB[buf][n * LDT + cq]) = pk.q;
        }
      }
    } else {
      // B[K][N] f32: transpose while staging (scalar ds stores)
#pragma unroll
      for (int cc = 0; cc < BNt / 64; ++cc) {
        const int slot = t + cc * 256;
        const int kk = slot / (BNt / 8);
        const int nq = (slot % (BNt / 8)) * 8;
#pragma unroll
        for (int j = 0; j < 8; ++j) {
          int gn = bn + nq + j;
          unsigned short hv = 0;
          if (gn < N && (k0 + kk) < K)
            hv = f2b(Bf[(size_t)(k0 + kk) * N + gn]);
          sB[buf][(nq + j) * LDT + kk] = hv;
        }
      }
    }
  };

  const int ksteps = (K + BK - 1) / BK;

  stage(0, 0);
  if (ASYNC_B) asm volatile("s_wait_asynccnt 0x0" ::: "memory");
  __syncthreads();

  for (int kt = 0; kt < ksteps; ++kt) {
    const int cur = kt & 1;
    if (kt + 1 < ksteps) stage(kt + 1, cur ^ 1);

    // ---- fragments per ISA bf16 A(16x32)/B(32x16) layouts ----
    const unsigned short* cA = sA[cur];
    const unsigned short* cB = sB[cur];
    FragBF fa[WR], fb[WC];
#pragma unroll
    for (int r = 0; r < WR; ++r) {
      const int arow = wm + r * 16 + ln;
      fa[r].q[0] = *reinterpret_cast<const uint4*>(&cA[arow * LDT + 8 * lh]);
      fa[r].q[1] = *reinterpret_cast<const uint4*>(&cA[arow * LDT + 16 + 8 * lh]);
    }
#pragma unroll
    for (int c = 0; c < WC; ++c) {
      const unsigned short* p = &cB[(wn + c * 16 + ln) * LDT];
      fb[c].q[0] = *reinterpret_cast<const uint4*>(p + 8 * lh);
      fb[c].q[1] = *reinterpret_cast<const uint4*>(p + 16 + 8 * lh);
    }
#pragma unroll
    for (int r = 0; r < WR; ++r)
#pragma unroll
      for (int c = 0; c < WC; ++c)
        acc[r][c] = __builtin_amdgcn_wmma_f32_16x16x32_bf16(
            false, fa[r].v, false, fb[c].v, (short)0, acc[r][c], false, false);

    if (ASYNC_B) asm volatile("s_wait_asynccnt 0x0" ::: "memory");
    __syncthreads();
  }

  // ---- epilogue: D element v -> (M = v + 8*lh, N = lane%16) ----
  int itv = 1;
  if (EPI == 2 && itp) itv = itp[0];
#pragma unroll
  for (int r = 0; r < WR; ++r) {
#pragma unroll
    for (int v = 0; v < 8; ++v) {
      int gm = bm + wm + r * 16 + v + 8 * lh;
      if (gm >= M) continue;
      int orow = gm;
      if (c_bsr) orow = c_base + (gm / c_rpb) * c_bsr + (gm % c_rpb);
      size_t rowoff = (size_t)orow * N;
#pragma unroll
      for (int c = 0; c < WC; ++c) {
        int gn = bn + wn + c * 16 + ln;
        if (gn >= N) continue;
        float a = acc[r][c][v];
        float outv;
        if (EPI == 1) {
          if (bias) a += bias[gn];
          outv = fmaxf(a, 0.f);
        } else if (EPI == 2) {
          float tt = (itv > 0) ? tanhf(a) : 0.f;
          outv = Rb[rowoff + gn] + tt;
        } else {
          outv = a;
        }
        Cb[rowoff + gn] = outv;
      }
    }
  }
}

// ---------------------------------------------------------------------------
// weight_norm(dim=0): W[o,k] = v[o,k] * g[o] / ||v[o,:]||, emitted as bf16.
// ---------------------------------------------------------------------------
__global__ __launch_bounds__(256) void wnorm_bf16(const float* __restrict__ v,
                                                  const float* __restrict__ g,
                                                  unsigned short* __restrict__ w,
                                                  int K)
{
  __shared__ float sm[256];
  const int o = blockIdx.x, t = threadIdx.x;
  float p = 0.f;
  for (int k = t; k < K; k += 256) { float x = v[(size_t)o * K + k]; p += x * x; }
  sm[t] = p; __syncthreads();
  for (int s = 128; s > 0; s >>= 1) { if (t < s) sm[t] += sm[t + s]; __syncthreads(); }
  float scale = g[o] * rsqrtf(sm[0]);
  __syncthreads();
  for (int k = t; k < K; k += 256)
    w[(size_t)o * K + k] = f2b(v[(size_t)o * K + k] * scale);
}

// concat(pi, ps) -> pcat [B, 150, 4]
__global__ void concat_pi_ps(const float* __restrict__ pi, const float* __restrict__ ps,
                             float* __restrict__ pcat)
{
  int idx = blockIdx.x * 256 + threadIdx.x;
  if (idx >= B_ * NTOT_ * 4) return;
  int c = idx & 3; int rn = idx >> 2; int n = rn % NTOT_; int b = rn / NTOT_;
  pcat[idx] = (n < LOC_) ? pi[(b * LOC_ + n) * 4 + c]
                         : ps[(b * NS_ + (n - LOC_)) * 4 + c];
}

// edges = concat(fea, bb) ; edgesL = edges * l_fa[b,:]
__global__ void build_edges(const float* __restrict__ fea, const float* __restrict__ bbf,
                            const float* __restrict__ lfa,
                            float* __restrict__ edges, float* __restrict__ edgesL)
{
  long long idx = (long long)blockIdx.x * 256 + threadIdx.x;
  const long long tot = (long long)B_ * NTOT_ * FDBB_;
  if (idx >= tot) return;
  int d = (int)(idx % FDBB_);
  long long r = idx / FDBB_;
  int b = (int)(r / NTOT_);
  float v = (d < FD_) ? fea[r * FD_ + d] : bbf[r * BBD_ + (d - FD_)];
  edges[idx] = v;
  edgesL[idx] = v * lfa[(size_t)b * FDBB_ + d];
}

// softmax along the feature dim of x [rows][ncols], in place
__global__ __launch_bounds__(256) void softmax_rows(float* __restrict__ x, int ncols)
{
  __shared__ float sm[256];
  const int t = threadIdx.x;
  float* row = x + (size_t)blockIdx.x * ncols;
  float mx = -INFINITY;
  for (int c = t; c < ncols; c += 256) mx = fmaxf(mx, row[c]);
  sm[t] = mx; __syncthreads();
  for (int s = 128; s > 0; s >>= 1) { if (t < s) sm[t] = fmaxf(sm[t], sm[t + s]); __syncthreads(); }
  mx = sm[0]; __syncthreads();
  float sum = 0.f;
  for (int c = t; c < ncols; c += 256) sum += __expf(row[c] - mx);
  sm[t] = sum; __syncthreads();
  for (int s = 128; s > 0; s >>= 1) { if (t < s) sm[t] += sm[t + s]; __syncthreads(); }
  float inv = 1.0f / sm[0];
  __syncthreads();
  for (int c = t; c < ncols; c += 256) row[c] = __expf(row[c] - mx) * inv;
}

// masked softmax over att[b][n][:], in place. allowed(m) = (n>=ms) || (m<ms)
__global__ __launch_bounds__(256) void att_softmax(float* __restrict__ att,
                                                   const int* __restrict__ mask_s)
{
  __shared__ float sm[256];
  const int t = threadIdx.x;
  const int n = blockIdx.x % NTOT_, b = blockIdx.x / NTOT_;
  const int ms = mask_s[b];
  const bool rowfree = (n >= ms);
  float* row = att + ((size_t)b * NTOT_ + n) * NTOT_;
  float mx = -INFINITY;
  for (int m = t; m < NTOT_; m += 256)
    if (rowfree || m < ms) mx = fmaxf(mx, row[m]);
  sm[t] = mx; __syncthreads();
  for (int s = 128; s > 0; s >>= 1) { if (t < s) sm[t] = fmaxf(sm[t], sm[t + s]); __syncthreads(); }
  mx = sm[0]; __syncthreads();
  float sum = 0.f;
  for (int m = t; m < NTOT_; m += 256)
    if (rowfree || m < ms) sum += __expf(row[m] - mx);
  sm[t] = sum; __syncthreads();
  for (int s = 128; s > 0; s >>= 1) { if (t < s) sm[t] += sm[t + s]; __syncthreads(); }
  float inv = 1.0f / sm[0];
  __syncthreads();
  for (int m = t; m < NTOT_; m += 256)
    row[m] = (rowfree || m < ms) ? __expf(row[m] - mx) * inv : 0.f;
}

// ---------------------------------------------------------------------------
// Host orchestration
// ---------------------------------------------------------------------------
static inline int cdiv(int a, int b) { return (a + b - 1) / b; }

template<bool BNN, bool BF32, int EPI, int BMt, int BNt>
static void run_gemm(hipStream_t st,
                     const float* A, const void* Bp, const float* bias,
                     const float* Rsd, const int* itp, float* C,
                     int M, int N, int K,
                     long long aBS, long long bBS, long long cBS,
                     int a_rpb, int a_bsr, int a_base,
                     int c_rpb, int c_bsr, int c_base, int batch)
{
  dim3 g(cdiv(N, BNt), cdiv(M, BMt), batch);
  gemm_wmma_bf16<BNN, BF32, EPI, BMt, BNt><<<g, dim3(256), 0, st>>>(
      A, Bp, bias, Rsd, itp, C, M, N, K, aBS, bBS, cBS,
      a_rpb, a_bsr, a_base, c_rpb, c_bsr, c_base);
}

extern "C" void kernel_launch(void* const* d_in, const int* in_sizes, int n_in,
                              void* d_out, int out_size, void* d_ws, size_t ws_size,
                              hipStream_t stream) {
  (void)in_sizes; (void)n_in; (void)out_size; (void)ws_size;

  const float* l    = (const float*)d_in[0];
  const float* iin  = (const float*)d_in[1];
  const float* s    = (const float*)d_in[2];
  const float* pi   = (const float*)d_in[3];
  const float* ps   = (const float*)d_in[4];
  const int*   msk  = (const int*)  d_in[5];
  const int*   itp  = (const int*)  d_in[6];
  const float* bb_v = (const float*)d_in[7];
  const float* bb_g = (const float*)d_in[8];
  const float* bb_b = (const float*)d_in[9];
  const float* i_v  = (const float*)d_in[10];
  const float* i_g  = (const float*)d_in[11];
  const float* i_b  = (const float*)d_in[12];
  const float* s_v  = (const float*)d_in[13];
  const float* s_g  = (const float*)d_in[14];
  const float* s_b  = (const float*)d_in[15];
  const float* l_v  = (const float*)d_in[16];
  const float* l_g  = (const float*)d_in[17];
  const float* l_b  = (const float*)d_in[18];
  const float* ir_v = (const float*)d_in[19];
  const float* ir_g = (const float*)d_in[20];
  const float* ir_b = (const float*)d_in[21];
  const float* sr_v = (const float*)d_in[22];
  const float* sr_g = (const float*)d_in[23];
  const float* sr_b = (const float*)d_in[24];

  // ---- workspace carve-out (256B aligned) ----
  char* ws = (char*)d_ws;
  size_t off = 0;
  auto carve = [&](size_t bytes) -> char* {
    char* p = ws + off;
    off += (bytes + 255) & ~(size_t)255;
    return p;
  };
  const int ROWS = B_ * NTOT_;                            // 38400
  unsigned short* Wbb = (unsigned short*)carve((size_t)BBD_ * 4    * 2);
  unsigned short* Wi  = (unsigned short*)carve((size_t)FD_  * DIN_ * 2);
  unsigned short* Ws  = (unsigned short*)carve((size_t)FD_  * SIN_ * 2);
  unsigned short* Wl  = (unsigned short*)carve((size_t)FDBB_* DIN_ * 2);
  unsigned short* Wir = (unsigned short*)carve((size_t)DIN_ * FD_  * 2);
  unsigned short* Wsr = (unsigned short*)carve((size_t)SIN_ * FD_  * 2);
  float* pcat  = (float*)carve((size_t)ROWS * 4     * 4);
  float* bbf   = (float*)carve((size_t)ROWS * BBD_  * 4);
  float* fea   = (float*)carve((size_t)ROWS * FD_   * 4);
  float* lfa   = (float*)carve((size_t)B_   * FDBB_ * 4);
  float* edges = (float*)carve((size_t)ROWS * FDBB_ * 4);
  float* edgesL= (float*)carve((size_t)ROWS * FDBB_ * 4);   // dead after att GEMM
  float* att   = (float*)carve((size_t)B_ * NTOT_ * NTOT_ * 4);
  float* fea2  = edgesL;                                    // alias onto dead edgesL

  float* out_i = (float*)d_out;                             // [B,100,2048]
  float* out_s = out_i + (size_t)B_ * LOC_ * DIN_;          // [B,50,300]

  // ---- 1) weight-norm all six matrices (fp32 -> bf16) ----
  wnorm_bf16<<<BBD_,  256, 0, stream>>>(bb_v, bb_g, Wbb, 4);
  wnorm_bf16<<<FD_,   256, 0, stream>>>(i_v,  i_g,  Wi,  DIN_);
  wnorm_bf16<<<FD_,   256, 0, stream>>>(s_v,  s_g,  Ws,  SIN_);
  wnorm_bf16<<<FDBB_, 256, 0, stream>>>(l_v,  l_g,  Wl,  DIN_);
  wnorm_bf16<<<DIN_,  256, 0, stream>>>(ir_v, ir_g, Wir, FD_);
  wnorm_bf16<<<SIN_,  256, 0, stream>>>(sr_v, sr_g, Wsr, FD_);

  // ---- 2) bb = relu(concat(pi,ps) @ Wbb^T + b)  [38400 x 64], K=4 ----
  concat_pi_ps<<<cdiv(B_ * NTOT_ * 4, 256), 256, 0, stream>>>(pi, ps, pcat);
  run_gemm<false, false, 1, 64, 64>(stream, pcat, Wbb, bb_b, nullptr, nullptr, bbf,
           ROWS, BBD_, 4, 0, 0, 0,
           ROWS, 0, 0, ROWS, 0, 0, 1);

  // ---- 3) i_fa -> fea rows [b*150 + 0..99] ; s_fa -> rows [b*150 + 100..149] ----
  run_gemm<false, false, 1, 128, 128>(stream, iin, Wi, i_b, nullptr, nullptr, fea,
           B_ * LOC_, FD_, DIN_, 0, 0, 0,
           B_ * LOC_, 0, 0, LOC_, NTOT_, 0, 1);
  run_gemm<false, false, 1, 128, 128>(stream, s, Ws, s_b, nullptr, nullptr, fea,
           B_ * NS_, FD_, SIN_, 0, 0, 0,
           B_ * NS_, 0, 0, NS_, NTOT_, LOC_, 1);

  // ---- 4) l_fa = softmax(relu(l @ Wl^T), axis=1) ----
  run_gemm<false, false, 1, 128, 128>(stream, l, Wl, l_b, nullptr, nullptr, lfa,
           B_, FDBB_, DIN_, 0, 0, 0,
           B_, 0, 0, B_, 0, 0, 1);
  softmax_rows<<<B_, 256, 0, stream>>>(lfa, FDBB_);

  // ---- 5) edges = concat(fea, bb); edgesL = edges * l_fa ----
  {
    long long tot = (long long)ROWS * FDBB_;
    build_edges<<<(unsigned)((tot + 255) / 256), 256, 0, stream>>>(fea, bbf, lfa,
                                                                   edges, edgesL);
  }

  // ---- 6) att[b] = edgesL[b] @ edges[b]^T  (batched NT, f32 src) ----
  run_gemm<false, true, 0, 64, 64>(stream, edgesL, edges, nullptr, nullptr, nullptr, att,
           NTOT_, NTOT_, FDBB_,
           (long long)NTOT_ * FDBB_, (long long)NTOT_ * FDBB_,
           (long long)NTOT_ * NTOT_,
           NTOT_, 0, 0, NTOT_, 0, 0, B_);

  // ---- 7) masked softmax along axis=2 ----
  att_softmax<<<B_ * NTOT_, 256, 0, stream>>>(att, msk);

  // ---- 8) fea2[b] = fea[b] + tanh(att[b] @ fea[b])  (batched NN, f32 src) ----
  // (fea is reset each reference iteration, so every it>=1 yields this result;
  //  the it scalar read in the epilogue gates the tanh term for it==0.)
  run_gemm<true, true, 2, 64, 64>(stream, att, fea, nullptr, fea, itp, fea2,
           NTOT_, FD_, NTOT_,
           (long long)NTOT_ * NTOT_, (long long)NTOT_ * FD_,
           (long long)NTOT_ * FD_,
           NTOT_, 0, 0, NTOT_, 0, 0, B_);

  // ---- 9) output heads: gather fea2 rows via A-remap ----
  run_gemm<false, false, 1, 128, 128>(stream, fea2, Wir, ir_b, nullptr, nullptr, out_i,
           B_ * LOC_, DIN_, FD_, 0, 0, 0,
           LOC_, NTOT_, 0, B_ * LOC_, 0, 0, 1);
  run_gemm<false, false, 1, 128, 128>(stream, fea2, Wsr, sr_b, nullptr, nullptr, out_s,
           B_ * NS_, SIN_, FD_, 0, 0, 0,
           NS_, NTOT_, LOC_, B_ * NS_, 0, 0, 1);
}
